// LSTMAutoencoderModel_50422916055443
// MI455X (gfx1250) — matrix operand order repaired and checked by reading, not compiled, not run on hardware
//
#include <hip/hip_runtime.h>

typedef __attribute__((ext_vector_type(16))) _Float16 v16h;
typedef __attribute__((ext_vector_type(8)))  float    v8f;

#define B_SZ   2048
#define T_SZ   128
#define FEAT   8
#define UO     128
#define UI     64
#define BN_EPS 1e-3f

#define WMMA_F16(ACC, AV, BV) \
  ACC = __builtin_amdgcn_wmma_f32_16x16x32_f16(false, AV, false, BV, (short)0, ACC, false, false)

__device__ __forceinline__ float sigmoid_f(float x) { return 1.0f / (1.0f + __expf(-x)); }
// branchless SELU: for x>0 the exp-term is exactly zero (exp(0)-1 == 0)
__device__ __forceinline__ float selu_f(float x) {
  const float a = 1.6732632423543772f, s = 1.0507009873554805f;
  return s * fmaxf(x, 0.f) + (s * a) * (__expf(fminf(x, 0.f)) - 1.f);
}

// B-matrix fragment (32x16, f16) for v_wmma_f32_16x16x32_f16, built from
// row-major f32 W[K][ldw]. Column n = col16 + lane%16; lane<16 holds K rows
// k0..k0+15, lane>=16 holds k0+16..k0+31 (one K value per half).
__device__ __forceinline__ v16h load_b_frag_f32(const float* __restrict__ W, int ldw,
                                                int k0, int col16, int lane) {
  int n  = col16 + (lane & 15);
  int kb = k0 + ((lane >> 4) << 4);
  v16h b;
#pragma unroll
  for (int j = 0; j < 16; ++j) b[j] = (_Float16)W[(size_t)(kb + j) * ldw + n];
  return b;
}

// A-matrix fragment (16x32, f16) from LDS row-major f16 buf[16][ldh].
// Lane m (m=lane%16): halves j=0..15 -> K=j; hi lanes +16.
__device__ __forceinline__ v16h load_a_frag_lds(const _Float16* buf, int ldh,
                                                int k0, int lane) {
  int m  = lane & 15;
  int kb = k0 + ((lane >> 4) << 4);
  const _Float16* row = buf + m * ldh + kb;
  v16h a;
#pragma unroll
  for (int j = 0; j < 16; ++j) a[j] = row[j];
  return a;
}

// A-matrix fragment directly from global f16 (one consumer wave, no LDS).
__device__ __forceinline__ v16h load_a_frag_glb(const _Float16* __restrict__ base, size_t ldr,
                                                int k0, int lane) {
  int m  = lane & 15;
  int kb = k0 + ((lane >> 4) << 4);
  const _Float16* row = base + (size_t)m * ldr + kb;
  v16h a;
#pragma unroll
  for (int j = 0; j < 16; ++j) a[j] = row[j];
  return a;
}

// ---------------- Stage 1: LSTM(F=8 -> U=128), seq out, BN folded -------------
__global__ void __launch_bounds__(256)
lstm1_kernel(const float* __restrict__ x,   // [B,T,8]
             const float* __restrict__ Wk,  // [8,512]
             const float* __restrict__ Wr,  // [128,512]
             const float* __restrict__ bias, const float* __restrict__ gam,
             const float* __restrict__ bet, const float* __restrict__ mmu,
             const float* __restrict__ vvar,
             _Float16* __restrict__ hout)   // [B,T,128] (BN applied)
{
  const int U = UO, NG = 4 * UO;
  __shared__ _Float16 hbuf[2][16 * UO];
  __shared__ float    xbuf[16 * FEAT];
  const int tid = threadIdx.x, lane = tid & 31, wave = tid >> 5;
  const int rb = blockIdx.x * 16;
  const int u0 = wave * 16, nl = lane & 15, col = u0 + nl;

  const float bi = bias[0 * U + col], bf2 = bias[1 * U + col];
  const float bg = bias[2 * U + col], bo = bias[3 * U + col];
  const float scl = gam[col] * rsqrtf(vvar[col] + BN_EPS);
  const float sht = bet[col] - mmu[col] * scl;

  float wkc[4][8];
#pragma unroll
  for (int g = 0; g < 4; ++g)
#pragma unroll
    for (int f = 0; f < 8; ++f) wkc[g][f] = Wk[f * NG + g * U + col];

  v16h wrf[4][4];
#pragma unroll
  for (int kc = 0; kc < 4; ++kc)
#pragma unroll
    for (int g = 0; g < 4; ++g)
      wrf[kc][g] = load_b_frag_f32(Wr, NG, kc * 32, g * U + u0, lane);

  for (int i = tid; i < 16 * U; i += blockDim.x) hbuf[0][i] = (_Float16)0.f;
  v8f c = {};

  // software-pipelined x-tile: registers hold tile for step t
  const int xr_r = tid >> 3, xr_f = tid & 7;  // valid for tid < 128
  float xreg = 0.f;
  if (tid < 16 * FEAT)
    xreg = x[((size_t)(rb + xr_r) * T_SZ + 0) * FEAT + xr_f];
  __syncthreads();

  for (int t = 0; t < T_SZ; ++t) {
    const int cur = t & 1, nxt = cur ^ 1;
    if (tid < 16 * FEAT) xbuf[tid] = xreg;
    if (t + 1 < T_SZ && tid < 16 * FEAT)
      xreg = x[((size_t)(rb + xr_r) * T_SZ + (t + 1)) * FEAT + xr_f];
    __syncthreads();

    v8f ai = {}, af = {}, ag = {}, ao = {};
#pragma unroll
    for (int kc = 0; kc < 4; ++kc) {
      v16h a = load_a_frag_lds(hbuf[cur], U, kc * 32, lane);
      WMMA_F16(ai, a, wrf[kc][0]);
      WMMA_F16(af, a, wrf[kc][1]);
      WMMA_F16(ag, a, wrf[kc][2]);
      WMMA_F16(ao, a, wrf[kc][3]);
    }
#pragma unroll
    for (int v = 0; v < 8; ++v) {
      const int m = v + ((lane >> 4) << 3);
      float zi = ai[v] + bi, zf = af[v] + bf2, zg = ag[v] + bg, zo = ao[v] + bo;
#pragma unroll
      for (int f = 0; f < 8; ++f) {
        const float xv = xbuf[m * 8 + f];
        zi += xv * wkc[0][f]; zf += xv * wkc[1][f];
        zg += xv * wkc[2][f]; zo += xv * wkc[3][f];
      }
      const float iv = sigmoid_f(zi), fv = sigmoid_f(zf);
      const float gv = selu_f(zg), ov = sigmoid_f(zo);
      const float cn = fv * c[v] + iv * gv;
      c[v] = cn;
      const float hn = ov * selu_f(cn);
      hbuf[nxt][m * U + col] = (_Float16)hn;
      hout[((size_t)(rb + m) * T_SZ + t) * UO + col] = (_Float16)(hn * scl + sht);
    }
    __syncthreads();
  }
}

// ---------------- Stage 2: LSTM(128 -> 64), last step only, BN folded --------
__global__ void __launch_bounds__(128)
lstm2_kernel(const _Float16* __restrict__ xin,  // [B,T,128]
             const float* __restrict__ Wk,      // [128,256]
             const float* __restrict__ Wr,      // [64,256]
             const float* __restrict__ bias, const float* __restrict__ gam,
             const float* __restrict__ bet, const float* __restrict__ mmu,
             const float* __restrict__ vvar,
             _Float16* __restrict__ enc)        // [B,64] (BN applied)
{
  const int U = UI, FIN = UO, NG = 4 * UI;
  const int NSTG = (16 * FIN / 2) / 128;  // uints per thread = 8
  __shared__ _Float16 hbuf[2][16 * UI];
  __shared__ _Float16 xbuf[16 * UO];
  const int tid = threadIdx.x, lane = tid & 31, wave = tid >> 5;
  const int rb = blockIdx.x * 16;
  const int u0 = wave * 16, nl = lane & 15, col = u0 + nl;

  const float bi = bias[0 * U + col], bf2 = bias[1 * U + col];
  const float bg = bias[2 * U + col], bo = bias[3 * U + col];
  const float scl = gam[col] * rsqrtf(vvar[col] + BN_EPS);
  const float sht = bet[col] - mmu[col] * scl;

  v16h wkf[4][4], wrf[2][4];
#pragma unroll
  for (int kc = 0; kc < 4; ++kc)
#pragma unroll
    for (int g = 0; g < 4; ++g)
      wkf[kc][g] = load_b_frag_f32(Wk, NG, kc * 32, g * U + u0, lane);
#pragma unroll
  for (int kc = 0; kc < 2; ++kc)
#pragma unroll
    for (int g = 0; g < 4; ++g)
      wrf[kc][g] = load_b_frag_f32(Wr, NG, kc * 32, g * U + u0, lane);

  for (int i = tid; i < 16 * U; i += blockDim.x) hbuf[0][i] = (_Float16)0.f;
  v8f c = {};

  // software-pipelined input tile in registers
  unsigned int stg[NSTG];
#pragma unroll
  for (int q = 0; q < NSTG; ++q) {
    const int i = tid + q * 128, r = i / (FIN / 2), cp = i % (FIN / 2);
    stg[q] = ((const unsigned int*)(xin + ((size_t)(rb + r) * T_SZ + 0) * FIN))[cp];
  }
  __syncthreads();

  for (int t = 0; t < T_SZ; ++t) {
    const int cur = t & 1, nxt = cur ^ 1;
#pragma unroll
    for (int q = 0; q < NSTG; ++q)
      ((unsigned int*)xbuf)[tid + q * 128] = stg[q];
    if (t + 1 < T_SZ) {
#pragma unroll
      for (int q = 0; q < NSTG; ++q) {
        const int i = tid + q * 128, r = i / (FIN / 2), cp = i % (FIN / 2);
        stg[q] = ((const unsigned int*)(xin + ((size_t)(rb + r) * T_SZ + (t + 1)) * FIN))[cp];
      }
    }
    __syncthreads();

    v8f ai = {}, af = {}, ag = {}, ao = {};
#pragma unroll
    for (int kc = 0; kc < 4; ++kc) {
      v16h a = load_a_frag_lds(xbuf, FIN, kc * 32, lane);
      WMMA_F16(ai, a, wkf[kc][0]);
      WMMA_F16(af, a, wkf[kc][1]);
      WMMA_F16(ag, a, wkf[kc][2]);
      WMMA_F16(ao, a, wkf[kc][3]);
    }
#pragma unroll
    for (int kc = 0; kc < 2; ++kc) {
      v16h a = load_a_frag_lds(hbuf[cur], U, kc * 32, lane);
      WMMA_F16(ai, a, wrf[kc][0]);
      WMMA_F16(af, a, wrf[kc][1]);
      WMMA_F16(ag, a, wrf[kc][2]);
      WMMA_F16(ao, a, wrf[kc][3]);
    }
#pragma unroll
    for (int v = 0; v < 8; ++v) {
      const int m = v + ((lane >> 4) << 3);
      const float iv = sigmoid_f(ai[v] + bi), fv = sigmoid_f(af[v] + bf2);
      const float gv = selu_f(ag[v] + bg), ov = sigmoid_f(ao[v] + bo);
      const float cn = fv * c[v] + iv * gv;
      c[v] = cn;
      hbuf[nxt][m * U + col] = (_Float16)(ov * selu_f(cn));
    }
    __syncthreads();
  }
  // final h is in hbuf[0] (T even); apply BN and emit encoder state
#pragma unroll
  for (int v = 0; v < 8; ++v) {
    const int m = v + ((lane >> 4) << 3);
    const float hn = (float)hbuf[0][m * U + col];
    enc[(size_t)(rb + m) * UI + col] = (_Float16)(hn * scl + sht);
  }
}

// ---------------- Stage 3: LSTM(64 -> 64) over repeated enc, seq out ---------
__global__ void __launch_bounds__(128)
lstm3_kernel(const _Float16* __restrict__ enc,  // [B,64]
             const float* __restrict__ Wk,      // [64,256]
             const float* __restrict__ Wr,      // [64,256]
             const float* __restrict__ bias, const float* __restrict__ gam,
             const float* __restrict__ bet, const float* __restrict__ mmu,
             const float* __restrict__ vvar,
             _Float16* __restrict__ dout)       // [B,T,64] (BN applied)
{
  const int U = UI, NG = 4 * UI;
  __shared__ _Float16 hbuf[2][16 * UI];
  __shared__ _Float16 xbuf[16 * UI];
  const int tid = threadIdx.x, lane = tid & 31, wave = tid >> 5;
  const int rb = blockIdx.x * 16;
  const int u0 = wave * 16, nl = lane & 15, col = u0 + nl;

  const float bi = bias[0 * U + col], bf2 = bias[1 * U + col];
  const float bg = bias[2 * U + col], bo = bias[3 * U + col];
  const float scl = gam[col] * rsqrtf(vvar[col] + BN_EPS);
  const float sht = bet[col] - mmu[col] * scl;

  v16h wrf[2][4];
#pragma unroll
  for (int kc = 0; kc < 2; ++kc)
#pragma unroll
    for (int g = 0; g < 4; ++g)
      wrf[kc][g] = load_b_frag_f32(Wr, NG, kc * 32, g * U + u0, lane);

  for (int i = tid; i < 16 * U; i += blockDim.x) hbuf[0][i] = (_Float16)0.f;
  for (int i = tid; i < 16 * U / 2; i += blockDim.x)
    ((unsigned int*)xbuf)[i] = ((const unsigned int*)(enc + (size_t)rb * U))[i];
  __syncthreads();

  // input is constant across time: hoist z_base = enc @ Wk out of the loop
  v8f zi = {}, zf = {}, zg = {}, zo = {};
#pragma unroll
  for (int kc = 0; kc < 2; ++kc) {
    v16h a = load_a_frag_lds(xbuf, U, kc * 32, lane);
    v16h b0 = load_b_frag_f32(Wk, NG, kc * 32, 0 * U + u0, lane);
    v16h b1 = load_b_frag_f32(Wk, NG, kc * 32, 1 * U + u0, lane);
    v16h b2 = load_b_frag_f32(Wk, NG, kc * 32, 2 * U + u0, lane);
    v16h b3 = load_b_frag_f32(Wk, NG, kc * 32, 3 * U + u0, lane);
    WMMA_F16(zi, a, b0);
    WMMA_F16(zf, a, b1);
    WMMA_F16(zg, a, b2);
    WMMA_F16(zo, a, b3);
  }

  v8f c = {};
  for (int t = 0; t < T_SZ; ++t) {
    const int cur = t & 1, nxt = cur ^ 1;
    v8f ai = zi, af = zf, ag = zg, ao = zo;
#pragma unroll
    for (int kc = 0; kc < 2; ++kc) {
      v16h a = load_a_frag_lds(hbuf[cur], U, kc * 32, lane);
      WMMA_F16(ai, a, wrf[kc][0]);
      WMMA_F16(af, a, wrf[kc][1]);
      WMMA_F16(ag, a, wrf[kc][2]);
      WMMA_F16(ao, a, wrf[kc][3]);
    }
#pragma unroll
    for (int v = 0; v < 8; ++v) {
      const int m = v + ((lane >> 4) << 3);
      const float iv = sigmoid_f(ai[v] + bi), fv = sigmoid_f(af[v] + bf2);
      const float gv = selu_f(ag[v] + bg), ov = sigmoid_f(ao[v] + bo);
      const float cn = fv * c[v] + iv * gv;
      c[v] = cn;
      const float hn = ov * selu_f(cn);
      hbuf[nxt][m * U + col] = (_Float16)hn;
      dout[((size_t)(rb + m) * T_SZ + t) * UI + col] = (_Float16)(hn * scl + sht);
    }
    __syncthreads();
  }
}

// ---------------- Stage 4: LSTM(64 -> 128), seq out, BN folded ---------------
__global__ void __launch_bounds__(256)
lstm4_kernel(const _Float16* __restrict__ xin,  // [B,T,64]
             const float* __restrict__ Wk,      // [64,512]
             const float* __restrict__ Wr,      // [128,512]
             const float* __restrict__ bias, const float* __restrict__ gam,
             const float* __restrict__ bet, const float* __restrict__ mmu,
             const float* __restrict__ vvar,
             _Float16* __restrict__ dout)       // [B,T,128] (BN applied)
{
  const int U = UO, FIN = UI, NG = 4 * UO;
  const int NSTG = (16 * FIN / 2) / 256;  // uints per thread = 2
  __shared__ _Float16 hbuf[2][16 * UO];
  __shared__ _Float16 xbuf[16 * UI];
  const int tid = threadIdx.x, lane = tid & 31, wave = tid >> 5;
  const int rb = blockIdx.x * 16;
  const int u0 = wave * 16, nl = lane & 15, col = u0 + nl;

  const float bi = bias[0 * U + col], bf2 = bias[1 * U + col];
  const float bg = bias[2 * U + col], bo = bias[3 * U + col];
  const float scl = gam[col] * rsqrtf(vvar[col] + BN_EPS);
  const float sht = bet[col] - mmu[col] * scl;

  v16h wkf[2][4], wrf[4][4];
#pragma unroll
  for (int kc = 0; kc < 2; ++kc)
#pragma unroll
    for (int g = 0; g < 4; ++g)
      wkf[kc][g] = load_b_frag_f32(Wk, NG, kc * 32, g * U + u0, lane);
#pragma unroll
  for (int kc = 0; kc < 4; ++kc)
#pragma unroll
    for (int g = 0; g < 4; ++g)
      wrf[kc][g] = load_b_frag_f32(Wr, NG, kc * 32, g * U + u0, lane);

  for (int i = tid; i < 16 * U; i += blockDim.x) hbuf[0][i] = (_Float16)0.f;
  v8f c = {};

  // software-pipelined input tile in registers
  unsigned int stg[NSTG];
#pragma unroll
  for (int q = 0; q < NSTG; ++q) {
    const int i = tid + q * 256, r = i / (FIN / 2), cp = i % (FIN / 2);
    stg[q] = ((const unsigned int*)(xin + ((size_t)(rb + r) * T_SZ + 0) * FIN))[cp];
  }
  __syncthreads();

  for (int t = 0; t < T_SZ; ++t) {
    const int cur = t & 1, nxt = cur ^ 1;
#pragma unroll
    for (int q = 0; q < NSTG; ++q)
      ((unsigned int*)xbuf)[tid + q * 256] = stg[q];
    if (t + 1 < T_SZ) {
#pragma unroll
      for (int q = 0; q < NSTG; ++q) {
        const int i = tid + q * 256, r = i / (FIN / 2), cp = i % (FIN / 2);
        stg[q] = ((const unsigned int*)(xin + ((size_t)(rb + r) * T_SZ + (t + 1)) * FIN))[cp];
      }
    }
    __syncthreads();

    v8f ai = {}, af = {}, ag = {}, ao = {};
#pragma unroll
    for (int kc = 0; kc < 2; ++kc) {
      v16h a = load_a_frag_lds(xbuf, FIN, kc * 32, lane);
      WMMA_F16(ai, a, wkf[kc][0]);
      WMMA_F16(af, a, wkf[kc][1]);
      WMMA_F16(ag, a, wkf[kc][2]);
      WMMA_F16(ao, a, wkf[kc][3]);
    }
#pragma unroll
    for (int kc = 0; kc < 4; ++kc) {
      v16h a = load_a_frag_lds(hbuf[cur], U, kc * 32, lane);
      WMMA_F16(ai, a, wrf[kc][0]);
      WMMA_F16(af, a, wrf[kc][1]);
      WMMA_F16(ag, a, wrf[kc][2]);
      WMMA_F16(ao, a, wrf[kc][3]);
    }
#pragma unroll
    for (int v = 0; v < 8; ++v) {
      const int m = v + ((lane >> 4) << 3);
      const float iv = sigmoid_f(ai[v] + bi), fv = sigmoid_f(af[v] + bf2);
      const float gv = selu_f(ag[v] + bg), ov = sigmoid_f(ao[v] + bo);
      const float cn = fv * c[v] + iv * gv;
      c[v] = cn;
      const float hn = ov * selu_f(cn);
      hbuf[nxt][m * U + col] = (_Float16)hn;
      dout[((size_t)(rb + m) * T_SZ + t) * UO + col] = (_Float16)(hn * scl + sht);
    }
    __syncthreads();
  }
}

// ---------------- Dense head: [B*T,128] @ [128,8] + bd ----------------------
__global__ void __launch_bounds__(256)
dense_kernel(const _Float16* __restrict__ d4,  // [B*T, 128] (BN already applied)
             const float* __restrict__ Wd,     // [128, 8]
             const float* __restrict__ bd,     // [8]
             float* __restrict__ out)          // [B*T, 8]
{
  const int lane = threadIdx.x & 31, wave = threadIdx.x >> 5;
  const size_t rowBase = ((size_t)blockIdx.x * 8 + wave) * 16;
  const int nl = lane & 15;

  v16h wf[4];
#pragma unroll
  for (int kc = 0; kc < 4; ++kc) {
    const int kb = kc * 32 + ((lane >> 4) << 4);
    v16h bfr;
#pragma unroll
    for (int j = 0; j < 16; ++j)
      bfr[j] = (nl < 8) ? (_Float16)Wd[(kb + j) * FEAT + nl] : (_Float16)0.f;
    wf[kc] = bfr;
  }

  v8f acc = {};
#pragma unroll
  for (int kc = 0; kc < 4; ++kc) {
    v16h a = load_a_frag_glb(d4 + rowBase * UO, UO, kc * 32, lane);
    WMMA_F16(acc, a, wf[kc]);
  }

  if (nl < 8) {
    const float bv = bd[nl];
#pragma unroll
    for (int v = 0; v < 8; ++v) {
      const int m = v + ((lane >> 4) << 3);
      out[(rowBase + m) * FEAT + nl] = acc[v] + bv;
    }
  }
}

extern "C" void kernel_launch(void* const* d_in, const int* in_sizes, int n_in,
                              void* d_out, int out_size, void* d_ws, size_t ws_size,
                              hipStream_t stream) {
  const float* x   = (const float*)d_in[0];
  const float* Wk1 = (const float*)d_in[1];  const float* Wr1 = (const float*)d_in[2];
  const float* b1  = (const float*)d_in[3];
  const float* g1  = (const float*)d_in[4];  const float* be1 = (const float*)d_in[5];
  const float* m1  = (const float*)d_in[6];  const float* v1  = (const float*)d_in[7];
  const float* Wk2 = (const float*)d_in[8];  const float* Wr2 = (const float*)d_in[9];
  const float* b2  = (const float*)d_in[10];
  const float* g2  = (const float*)d_in[11]; const float* be2 = (const float*)d_in[12];
  const float* m2  = (const float*)d_in[13]; const float* v2  = (const float*)d_in[14];
  const float* Wk3 = (const float*)d_in[15]; const float* Wr3 = (const float*)d_in[16];
  const float* b3  = (const float*)d_in[17];
  const float* g3  = (const float*)d_in[18]; const float* be3 = (const float*)d_in[19];
  const float* m3  = (const float*)d_in[20]; const float* v3  = (const float*)d_in[21];
  const float* Wk4 = (const float*)d_in[22]; const float* Wr4 = (const float*)d_in[23];
  const float* b4  = (const float*)d_in[24];
  const float* g4  = (const float*)d_in[25]; const float* be4 = (const float*)d_in[26];
  const float* m4  = (const float*)d_in[27]; const float* v4  = (const float*)d_in[28];
  const float* Wd  = (const float*)d_in[29]; const float* bd  = (const float*)d_in[30];

  char* ws = (char*)d_ws;
  _Float16* h1  = (_Float16*)(ws);                           // B*T*128 f16 = 64 MB
  _Float16* d3  = (_Float16*)(ws + ((size_t)64 << 20));      // B*T*64  f16 = 32 MB
  _Float16* d4  = (_Float16*)(ws + ((size_t)96 << 20));      // B*T*128 f16 = 64 MB
  _Float16* enc = (_Float16*)(ws + ((size_t)160 << 20));     // B*64    f16 = 256 KB

  const int nblk = B_SZ / 16;  // 128 batch tiles
  lstm1_kernel<<<nblk, 256, 0, stream>>>(x, Wk1, Wr1, b1, g1, be1, m1, v1, h1);
  lstm2_kernel<<<nblk, 128, 0, stream>>>(h1, Wk2, Wr2, b2, g2, be2, m2, v2, enc);
  lstm3_kernel<<<nblk, 128, 0, stream>>>(enc, Wk3, Wr3, b3, g3, be3, m3, v3, d3);
  lstm4_kernel<<<nblk, 256, 0, stream>>>(d3, Wk4, Wr4, b4, g4, be4, m4, v4, d4);
  dense_kernel<<<(B_SZ * T_SZ / 16) / 8, 256, 0, stream>>>(d4, Wd, bd, (float*)d_out);
}
